// LocalReverseDiffusion_72335839199839
// MI455X (gfx1250) — matrix operand order repaired and checked by reading, not compiled, not run on hardware
//
#include <hip/hip_runtime.h>
#include <hip/hip_bf16.h>

typedef __attribute__((ext_vector_type(16))) _Float16 v16h;
typedef __attribute__((ext_vector_type(8)))  _Float16 v8h;
typedef __attribute__((ext_vector_type(8)))  float    v8f;
typedef __attribute__((ext_vector_type(4)))  float    v4f;

#define NC      64
#define DHW     4096      // 16^3 input spatial
#define OUT_SP  262144    // 64^3 output spatial

// ---------------------------------------------------------------------------
// Prep 1: x [n,c,p] f32  ->  xT [n,p,c] f16  (c contiguous => B-fragment B128 loads)
// ---------------------------------------------------------------------------
__global__ void convert_x_kernel(const float* __restrict__ x, _Float16* __restrict__ xT) {
    int idx = blockIdx.x * blockDim.x + threadIdx.x;           // 2*64*4096 = 524288
    int p = idx & (DHW - 1);
    int c = (idx >> 12) & 63;
    int n = idx >> 18;
    xT[(n * DHW + p) * NC + c] = (_Float16)x[idx];
}

// ---------------------------------------------------------------------------
// Prep 2: Wtab[ijk][o][c] = w_pw[o,c] * gamma[c] * w_ct[c,ijk]   (f16, 512 KB)
// ---------------------------------------------------------------------------
__global__ void build_wtab_kernel(const float* __restrict__ w_pw,
                                  const float* __restrict__ gamma,
                                  const float* __restrict__ w_ct,
                                  _Float16* __restrict__ Wtab) {
    int idx = blockIdx.x * blockDim.x + threadIdx.x;           // 64*64*64 = 262144
    int c   = idx & 63;
    int o   = (idx >> 6) & 63;
    int ijk = idx >> 12;
    Wtab[idx] = (_Float16)(w_pw[o * NC + c] * gamma[c] * w_ct[c * 64 + ijk]);
}

// ---------------------------------------------------------------------------
// Prep 3: per-sample GroupNorm stats (analytic, never materializing y) and the
// folded per-output-channel constant K_o.
//   stats[n]            = rstd_n
//   stats[8 + n*64 + o] = K_o(n)
// ---------------------------------------------------------------------------
__global__ void stats_kernel(const float* __restrict__ x,
                             const float* __restrict__ w_ct,
                             const float* __restrict__ b_ct,
                             const float* __restrict__ gamma,
                             const float* __restrict__ beta,
                             const float* __restrict__ w_pw,
                             float* __restrict__ stats) {
    __shared__ float red[256], red2[256];
    __shared__ float sy_c[64], sy2_c[64];
    __shared__ float sMean, sRstd;

    const int n = blockIdx.x;
    const int t = threadIdx.x;
    const int c = t & 63;
    const int q = t >> 6;                                      // 4 chunks of 1024

    const float* xp = x + ((n * NC + c) * DHW) + q * 1024;
    float s = 0.f, s2 = 0.f;
    for (int p = 0; p < 1024; ++p) { float v = xp[p]; s += v; s2 += v * v; }
    red[t] = s; red2[t] = s2;
    __syncthreads();

    if (t < 64) {
        float Sx  = red[t]  + red[t + 64]  + red[t + 128]  + red[t + 192];
        float Sx2 = red2[t] + red2[t + 64] + red2[t + 128] + red2[t + 192];
        float Sw = 0.f, Sw2 = 0.f;
        for (int k = 0; k < 64; ++k) { float wv = w_ct[t * 64 + k]; Sw += wv; Sw2 += wv * wv; }
        float b = b_ct[t];
        sy_c[t]  = Sx * Sw + 262144.0f * b;
        sy2_c[t] = Sx2 * Sw2 + 2.0f * b * Sx * Sw + 262144.0f * b * b;
    }
    __syncthreads();

    if (t == 0) {
        float sy = 0.f, sy2 = 0.f;
        for (int cc = 0; cc < 64; ++cc) { sy += sy_c[cc]; sy2 += sy2_c[cc]; }
        const float Ntot = 64.0f * 262144.0f;
        float mean = sy / Ntot;
        float var  = sy2 / Ntot - mean * mean;
        float rstd = rsqrtf(var + 1e-5f);
        stats[n] = rstd;
        sMean = mean; sRstd = rstd;
    }
    __syncthreads();

    if (t < 64) {                                              // o = t
        float mean = sMean, rstd = sRstd;
        float acc = 0.f;
        for (int cc = 0; cc < 64; ++cc)
            acc += w_pw[t * NC + cc] * (gamma[cc] * (b_ct[cc] - mean) * rstd + beta[cc]);
        stats[8 + n * 64 + t] = acc;
    }
}

// ---------------------------------------------------------------------------
// Main fused kernel: per wave -> (n, i, j, o-tile, 4 spatial rows of 16).
// D = rstd * (Wtab_ijk[16x64] @ xT[64x16]) + K_o, K-dim = channels (2 WMMA
// chunks of 32). Four k-variants share B; results interleave into float4
// stores along w_out = 4*w + k (fully coalesced 256B per 16 lanes).
// ---------------------------------------------------------------------------
__global__ __launch_bounds__(256, 1) void fused_main_kernel(
    const _Float16* __restrict__ xT,
    const _Float16* __restrict__ Wtab,
    const float*    __restrict__ stats,
    float*          __restrict__ out) {

    const int lane = threadIdx.x & 31;
    const int hf   = lane >> 4;         // 0 | 1
    const int l16  = lane & 15;

    int gw = blockIdx.x * 8 + (threadIdx.x >> 5);   // 8192 waves total
    const int rg = gw & 63;  gw >>= 6;              // 64 groups of 4 rows
    const int ot = gw & 3;   gw >>= 2;              // o tile (x16)
    const int ij = gw & 15;  gw >>= 4;              // i*4 + j
    const int n  = gw & 1;
    const int i = ij >> 2, j = ij & 3;
    const int o_base = ot * 16;

    union V16 { v16h v; v8h h[2]; };

    // ---- A fragments: 4 output-k variants x 2 K-chunks, register resident.
    // 16-bit A 16x32 layout: lane(l16)=row M; elems 0..7 = K 8*hf..8*hf+7,
    // elems 8..15 = K 16+8*hf.. => two contiguous v8h runs in c.
    V16 A[4][2];
    const int o_lane = o_base + l16;
#pragma unroll
    for (int k = 0; k < 4; ++k) {
        const int ijk = i * 16 + j * 4 + k;
        const _Float16* Ap = Wtab + (ijk * 64 + o_lane) * 64;
#pragma unroll
        for (int kk = 0; kk < 2; ++kk) {
            A[k][kk].h[0] = *(const v8h*)(Ap + 32 * kk + 8 * hf);
            A[k][kk].h[1] = *(const v8h*)(Ap + 32 * kk + 16 + 8 * hf);
        }
    }

    const float rstd = stats[n];
    const float* kp = stats + 8 + n * 64 + o_base + 8 * hf;
    v4f kc0 = *(const v4f*)(kp);
    v4f kc1 = *(const v4f*)(kp + 4);
    float Kc[8];
#pragma unroll
    for (int r = 0; r < 4; ++r) { Kc[r] = kc0[r]; Kc[r + 4] = kc1[r]; }

#pragma unroll
    for (int rr = 0; rr < 4; ++rr) {
        const int row    = rg * 4 + rr;       // 0..255 flat (d,h) row
        const int p_base = row * 16;          // w = l16
        const int d = p_base >> 8;
        const int h = (p_base >> 4) & 15;

        // B fragments: 32x16 f16, lane(l16)=col N; elems 0..7 = K 8*hf..8*hf+7,
        // elems 8..15 = K 16+8*hf.., contiguous in c thanks to the xT[n,p,c] transpose.
        const _Float16* Bp = xT + (n * DHW + p_base + l16) * NC;
        V16 B0, B1;
        B0.h[0] = *(const v8h*)(Bp + 8 * hf);
        B0.h[1] = *(const v8h*)(Bp + 16 + 8 * hf);
        B1.h[0] = *(const v8h*)(Bp + 32 + 8 * hf);
        B1.h[1] = *(const v8h*)(Bp + 32 + 16 + 8 * hf);

        v8f acc[4];
#pragma unroll
        for (int k = 0; k < 4; ++k) {
            v8f a = {};
            a = __builtin_amdgcn_wmma_f32_16x16x32_f16(false, A[k][0].v, false, B0.v,
                                                       (short)0, a, false, false);
            a = __builtin_amdgcn_wmma_f32_16x16x32_f16(false, A[k][1].v, false, B1.v,
                                                       (short)0, a, false, false);
            acc[k] = a;
        }

        const int sp_off = (4 * d + i) * 4096 + (4 * h + j) * 64 + 4 * l16;
#pragma unroll
        for (int r = 0; r < 8; ++r) {
            const int o = o_base + r + 8 * hf;                 // D row M = r + 8*hf
            const int off = (n * NC + o) * OUT_SP + sp_off;
            v4f val;
            val.x = rstd * acc[0][r] + Kc[r];
            val.y = rstd * acc[1][r] + Kc[r];
            val.z = rstd * acc[2][r] + Kc[r];
            val.w = rstd * acc[3][r] + Kc[r];
            __builtin_nontemporal_store(val, (v4f*)(out + off));   // streaming 128MB
        }
    }
}

// ---------------------------------------------------------------------------
extern "C" void kernel_launch(void* const* d_in, const int* in_sizes, int n_in,
                              void* d_out, int out_size, void* d_ws, size_t ws_size,
                              hipStream_t stream) {
    const float* x     = (const float*)d_in[0];
    const float* w_ct  = (const float*)d_in[1];
    const float* b_ct  = (const float*)d_in[2];
    const float* gamma = (const float*)d_in[3];
    const float* beta  = (const float*)d_in[4];
    const float* w_pw  = (const float*)d_in[5];
    float* out = (float*)d_out;

    // ws layout: xT f16 [2*4096*64] | Wtab f16 [64*64*64] | stats f32 [136]
    _Float16* xT   = (_Float16*)d_ws;
    _Float16* Wtab = xT + 2 * DHW * NC;
    float*    stats = (float*)(Wtab + 64 * 64 * 64);

    convert_x_kernel<<<2048, 256, 0, stream>>>(x, xT);
    build_wtab_kernel<<<1024, 256, 0, stream>>>(w_pw, gamma, w_ct, Wtab);
    stats_kernel<<<2, 256, 0, stream>>>(x, w_ct, b_ct, gamma, beta, w_pw, stats);
    fused_main_kernel<<<1024, 256, 0, stream>>>(xT, Wtab, stats, out);
}
